// THNN_global_layer_90185723281673
// MI455X (gfx1250) — matrix operand order repaired and checked by reading, not compiled, not run on hardware
//
#include <hip/hip_runtime.h>
#include <hip/hip_bf16.h>

// ---------------------------------------------------------------------------
// Types for CDNA5 WMMA (wave32): v_wmma_f32_16x16x32_bf16
// ---------------------------------------------------------------------------
typedef __attribute__((ext_vector_type(16))) __bf16          v16bf;
typedef __attribute__((ext_vector_type(8)))  float           v8f;
typedef __attribute__((ext_vector_type(8)))  unsigned short  ush8;

union BFrag { v16bf v; ush8 h[2]; };

__device__ __forceinline__ v8f wmma_bf16(v16bf a, v16bf b, v8f c) {
  // 8 args: (neg_a, A, neg_b, B, c_mod, C, reuse_a, reuse_b)
  return __builtin_amdgcn_wmma_f32_16x16x32_bf16(false, a, false, b,
                                                 (short)0, c, false, false);
}

__device__ __forceinline__ unsigned short f2bf(float f) {
  unsigned u = __float_as_uint(f);
  unsigned r = u + 0x7FFFu + ((u >> 16) & 1u);  // round-to-nearest-even
  return (unsigned short)(r >> 16);
}

// A fragment (16x32 bf16, row m = lane&15) from an LDS tile with row pitch ldk.
// ISA layout: lanes 0-15 hold K = {k0..k0+7, k0+16..k0+23},
//             lanes 16-31 hold K = {k0+8..k0+15, k0+24..k0+31}.
__device__ __forceinline__ v16bf load_a_lds(const unsigned short* s, int ldk,
                                            int k0, int lane) {
  int row = lane & 15, lh = (lane >> 4) & 1;
  BFrag f;
  f.h[0] = *(const ush8*)(s + row * ldk + k0 + lh * 8);
  f.h[1] = *(const ush8*)(s + row * ldk + k0 + 16 + lh * 8);
  return f.v;
}

// B fragment (32x16 bf16, col n = n0 + (lane&15)) from transposed weight
// Wt[n][k] (n-major, pitch K). Lane holds K = {k0+8*lh..+7, k0+16+8*lh..+7}
// (same order as the A fragment).
__device__ __forceinline__ v16bf load_b_glb(const unsigned short* Wt, int n0,
                                            int K, int k0, int lane) {
  int n = n0 + (lane & 15), lh = (lane >> 4) & 1;
  const ush8* p = (const ush8*)(Wt + (size_t)n * K + k0 + lh * 8);
  BFrag f;
  f.h[0] = p[0];
  f.h[1] = p[2];
  return f.v;
}

// Unguarded 16x16 accumulator store (C layout: lane n, VGPR r -> row r+8*lh).
__device__ __forceinline__ void store_tile16(float* C, int ldc, int row0,
                                             int n0, const v8f& acc,
                                             const float* bias, bool do_relu,
                                             int lane) {
  int n  = n0 + (lane & 15);
  int lh = (lane >> 4) & 1;
  float b = bias[n];
#pragma unroll
  for (int r = 0; r < 8; ++r) {
    float v = acc[r] + b;
    if (do_relu) v = fmaxf(v, 0.0f);
    C[(size_t)(row0 + r + 8 * lh) * ldc + n] = v;
  }
}

// Row-guarded variant (tail tiles only).
__device__ __forceinline__ void store_tile_g(float* C, int ldc, int row0,
                                             int nrows, int n0, const v8f& acc,
                                             const float* bias, bool do_relu,
                                             int lane) {
  int n  = n0 + (lane & 15);
  int lh = (lane >> 4) & 1;
  float b = bias[n];
#pragma unroll
  for (int r = 0; r < 8; ++r) {
    int m = r + 8 * lh;
    if (m < nrows) {
      float v = acc[r] + b;
      if (do_relu) v = fmaxf(v, 0.0f);
      C[(size_t)(row0 + m) * ldc + n] = v;
    }
  }
}

// ---------------------------------------------------------------------------
// Prep kernels
// ---------------------------------------------------------------------------
__global__ __launch_bounds__(256) void k_cvt_bf16(const float* __restrict__ x,
                                                  unsigned short* __restrict__ y,
                                                  long long n) {
  for (long long i = (long long)blockIdx.x * blockDim.x + threadIdx.x; i < n;
       i += (long long)gridDim.x * blockDim.x)
    y[i] = f2bf(x[i]);
}

// W is (K [+1]) x Nout fp32 row-major.  Produces Wt[n*K + k] = bf16(W[k][n])
// and biasOut[n] = b[n] + (extraRow ? W[K][n] : 0)  (folds the ones-column).
__global__ __launch_bounds__(256) void k_prep_w(const float* __restrict__ W,
                                                const float* __restrict__ b,
                                                unsigned short* __restrict__ Wt,
                                                float* __restrict__ biasOut,
                                                int K, int Nout, int extraRow) {
  long long total = (long long)K * Nout;
  long long tid0  = (long long)blockIdx.x * blockDim.x + threadIdx.x;
  for (long long i = tid0; i < total; i += (long long)gridDim.x * blockDim.x) {
    int n = (int)(i / K), k = (int)(i % K);
    Wt[i] = f2bf(W[(size_t)k * Nout + n]);
  }
  if (tid0 < Nout)
    biasOut[tid0] = b[tid0] + (extraRow ? W[(size_t)K * Nout + tid0] : 0.0f);
}

__global__ __launch_bounds__(256) void k_zero(float* __restrict__ p, long long n) {
  for (long long i = (long long)blockIdx.x * blockDim.x + threadIdx.x; i < n;
       i += (long long)gridDim.x * blockDim.x)
    p[i] = 0.0f;
}

// ---------------------------------------------------------------------------
// Kernel 1: all node-level GEMMs, fused per 32-node tile (dual row-groups
// share each B fragment: 1 B load + 2 A loads + 2 WMMAs per K-step).
//   emb_new  = emb @ pW'   + pB               (N x 64)
//   residual = relu(emb @ aW' + aB)           (N x 256)
//   hid      = relu(emb @ p2W1' + hB)  -> LDS (32 x 1024 bf16)
//   emb_new2 = hid @ p2W2' + b2               (N x 256)
// Block: 256 threads = 8 waves; dynamic LDS: 16KB (sA) + 64KB (sH).
// ---------------------------------------------------------------------------
__global__ __launch_bounds__(256) void k_node(
    const unsigned short* __restrict__ embB,
    const unsigned short* __restrict__ aWt, const float* __restrict__ aB,
    const unsigned short* __restrict__ pWt, const float* __restrict__ pB,
    const unsigned short* __restrict__ h1Wt, const float* __restrict__ hB,
    const unsigned short* __restrict__ h2Wt, const float* __restrict__ b2,
    float* __restrict__ emb_new, float* __restrict__ residual,
    float* __restrict__ emb_new2, int N) {
  extern __shared__ unsigned short smem[];
  unsigned short* sA = smem;             // 32 x 256 bf16 emb tile
  unsigned short* sH = smem + 32 * 256;  // 32 x 1024 bf16 relu(hidden) tile

  const int row0 = blockIdx.x * 32;
  const int tid  = threadIdx.x;
  const int w = tid >> 5, lane = tid & 31, lh = (lane >> 4) & 1;
  const bool full = (row0 + 32 <= N);

  __builtin_prefetch(h1Wt, 0, 3);
  __builtin_prefetch(h2Wt, 0, 3);
  __builtin_prefetch(aWt, 0, 3);

  for (int i = tid; i < 32 * 256; i += 256) {
    int m = i >> 8, k = i & 255;
    int gr = row0 + m;
    sA[i] = (gr < N) ? embB[(size_t)gr * 256 + k] : (unsigned short)0;
  }
  __syncthreads();

  // ---- emb_new (N x 64): 8 tiles, one per wave (rg = w&1, col = w>>1) ----
  {
    int rg = w & 1, n0 = (w >> 1) * 16;
    v8f acc = {};
#pragma unroll
    for (int kk = 0; kk < 256; kk += 32)
      acc = wmma_bf16(load_a_lds(sA + rg * 16 * 256, 256, kk, lane),
                      load_b_glb(pWt, n0, 256, kk, lane), acc);
    int r0 = row0 + rg * 16;
    if (full)
      store_tile16(emb_new, 64, r0, n0, acc, pB, false, lane);
    else {
      int nr = N - r0;
      if (nr > 0) store_tile_g(emb_new, 64, r0, min(nr, 16), n0, acc, pB,
                               false, lane);
    }
  }

  // ---- residual (N x 256): cols {w, w+8}, dual row-group, K=256 ----
  for (int t = 0; t < 2; ++t) {
    int n0 = (w + 8 * t) * 16;
    v8f a0 = {}, a1 = {};
#pragma unroll
    for (int kk = 0; kk < 256; kk += 32) {
      v16bf b = load_b_glb(aWt, n0, 256, kk, lane);
      a0 = wmma_bf16(load_a_lds(sA, 256, kk, lane), b, a0);
      a1 = wmma_bf16(load_a_lds(sA + 16 * 256, 256, kk, lane), b, a1);
    }
    if (full) {
      store_tile16(residual, 256, row0, n0, a0, aB, true, lane);
      store_tile16(residual, 256, row0 + 16, n0, a1, aB, true, lane);
    } else {
      int nr = N - row0;
      if (nr > 0)
        store_tile_g(residual, 256, row0, min(nr, 16), n0, a0, aB, true, lane);
      if (nr > 16)
        store_tile_g(residual, 256, row0 + 16, nr - 16, n0, a1, aB, true, lane);
    }
  }

  // ---- hidden (32 x 1024) -> relu -> bf16 -> LDS; cols w + 8t ----
  for (int t = 0; t < 8; ++t) {
    int n0 = (w + 8 * t) * 16;
    v8f a0 = {}, a1 = {};
#pragma unroll
    for (int kk = 0; kk < 256; kk += 32) {
      v16bf b = load_b_glb(h1Wt, n0, 256, kk, lane);
      a0 = wmma_bf16(load_a_lds(sA, 256, kk, lane), b, a0);
      a1 = wmma_bf16(load_a_lds(sA + 16 * 256, 256, kk, lane), b, a1);
    }
    int n = n0 + (lane & 15);
    float bb = hB[n];
#pragma unroll
    for (int r = 0; r < 8; ++r) {
      int m = r + 8 * lh;
      sH[m * 1024 + n]        = f2bf(fmaxf(a0[r] + bb, 0.0f));
      sH[(m + 16) * 1024 + n] = f2bf(fmaxf(a1[r] + bb, 0.0f));
    }
  }
  __syncthreads();

  // ---- emb_new2 (N x 256) = sH @ p2W2' + b2; cols {w, w+8}, K=1024 ----
  for (int t = 0; t < 2; ++t) {
    int n0 = (w + 8 * t) * 16;
    v8f a0 = {}, a1 = {};
    for (int kk = 0; kk < 1024; kk += 32) {
      v16bf b = load_b_glb(h2Wt, n0, 1024, kk, lane);
      a0 = wmma_bf16(load_a_lds(sH, 1024, kk, lane), b, a0);
      a1 = wmma_bf16(load_a_lds(sH + 16 * 1024, 1024, kk, lane), b, a1);
    }
    if (full) {
      store_tile16(emb_new2, 256, row0, n0, a0, b2, false, lane);
      store_tile16(emb_new2, 256, row0 + 16, n0, a1, b2, false, lane);
    } else {
      int nr = N - row0;
      if (nr > 0)
        store_tile_g(emb_new2, 256, row0, min(nr, 16), n0, a0, b2, false, lane);
      if (nr > 16)
        store_tile_g(emb_new2, 256, row0 + 16, nr - 16, n0, a1, b2, false,
                     lane);
    }
  }
}

// ---------------------------------------------------------------------------
// Kernel 2: per-incidence work. One block = 64 incidence entries.
// Phase 1 (VALU): ee[m][d] = tanh(loo * gfac * inv_fact) -> bf16 in LDS.
// Phase 2 (WMMA): q = ee @ qW' (64 x 256, K=64); one B fragment feeds all
//                 4 row-groups.
// Phase 3: contrib = q + qb + relu(sum_j emb_new2[node_j]); atomic segsum.
// ---------------------------------------------------------------------------
__global__ __launch_bounds__(256) void k_edge(
    const float* __restrict__ emb_new, const float* __restrict__ emb_new2,
    const unsigned short* __restrict__ qWt, const float* __restrict__ qB,
    const float* __restrict__ global_emb, const int* __restrict__ edge_nodes,
    const int* __restrict__ edge_size, const int* __restrict__ node_degree,
    const int* __restrict__ inc_node, const int* __restrict__ inc_edge,
    const int* __restrict__ inc_slot, float* __restrict__ node_acc, int nnz) {
  __shared__ unsigned short sEE[64 * 64];  // 8 KB bf16 tanh(ee) tile
  __shared__ int   sNodes[64 * 4];
  __shared__ float sScale[64 * 4];
  __shared__ int   sSize[64];
  __shared__ int   sSlot[64];
  __shared__ int   sTgt[64];
  __shared__ float sInvf[64];

  const int i0  = blockIdx.x * 64;
  const int tid = threadIdx.x;
  const bool bfull = (i0 + 64 <= nnz);

  __builtin_prefetch(qWt, 0, 3);

  // --- Phase 0: per-entry metadata; thread -> (entry m = tid>>2, j = tid&3)
  {
    int m = tid >> 2, j = tid & 3;
    int idx = i0 + m;
    int nd = 0;
    float sc = 0.0f;
    if (bfull || idx < nnz) {
      int e = inc_edge[idx];
      int s = edge_size[e];
      float a = (s == 1) ? 1.0f : ((s <= 3) ? (1.0f / 3.0f) : 0.25f);
      nd = edge_nodes[(size_t)e * 4 + j];
      sc = powf((float)node_degree[nd], a);
      if (j == 0) {
        sSize[m] = s;
        sSlot[m] = inc_slot[idx];
        sTgt[m]  = inc_node[idx];
        sInvf[m] = (s <= 2) ? 1.0f : ((s == 3) ? 0.5f : (1.0f / 6.0f));
      }
    } else if (j == 0) {
      sSize[m] = 0;
      sSlot[m] = 0;
      sTgt[m]  = 0;
      sInvf[m] = 1.0f;
    }
    sNodes[m * 4 + j] = nd;
    sScale[m * 4 + j] = sc;
  }
  __syncthreads();

  // --- Phase 1: ee tile (64 entries x 64 dims) ---
  {
    int d  = tid & 63;
    int mb = tid >> 6;  // 0..3
    float ge = global_emb[d];
#pragma unroll 4
    for (int i = 0; i < 16; ++i) {
      int m = mb + 4 * i;
      unsigned short out = 0;
      if (bfull || (i0 + m) < nnz) {
        int s = sSize[m], slot = sSlot[m];
        float prod = 1.0f;
#pragma unroll
        for (int j = 0; j < 4; ++j) {
          if (j == slot) continue;
          float t = 1.0f;
          if (j < s)
            t = sScale[m * 4 + j] *
                emb_new[(size_t)sNodes[m * 4 + j] * 64 + d];
          prod *= t;
        }
        int kdeg = 4 - s;
        float g = 1.0f;
        for (int q = 0; q < kdeg; ++q) g *= ge;
        out = f2bf(tanhf(prod * g * sInvf[m]));
      }
      sEE[m * 64 + d] = out;
    }
  }
  __syncthreads();

  // --- Phase 2+3: q = ee @ qW' (K=64); cols {w, w+8}; 4 row-groups / B ---
  const int w = tid >> 5, lane = tid & 31, lh = (lane >> 4) & 1;

  for (int t = 0; t < 2; ++t) {
    int n0 = (w + 8 * t) * 16;
    v8f a0 = {}, a1 = {}, a2 = {}, a3 = {};
#pragma unroll
    for (int kk = 0; kk < 64; kk += 32) {
      v16bf b = load_b_glb(qWt, n0, 64, kk, lane);
      a0 = wmma_bf16(load_a_lds(sEE + 0 * 16 * 64, 64, kk, lane), b, a0);
      a1 = wmma_bf16(load_a_lds(sEE + 1 * 16 * 64, 64, kk, lane), b, a1);
      a2 = wmma_bf16(load_a_lds(sEE + 2 * 16 * 64, 64, kk, lane), b, a2);
      a3 = wmma_bf16(load_a_lds(sEE + 3 * 16 * 64, 64, kk, lane), b, a3);
    }

    int n = n0 + (lane & 15);
    float qbn = qB[n];
    const v8f* accs[4] = {&a0, &a1, &a2, &a3};
#pragma unroll
    for (int rg = 0; rg < 4; ++rg) {
      const v8f& acc = *accs[rg];
#pragma unroll
      for (int r = 0; r < 8; ++r) {
        int m = rg * 16 + r + 8 * lh;
        if (bfull || (i0 + m) < nnz) {
          int s = sSize[m];
          float es = 0.0f;
#pragma unroll
          for (int j = 0; j < 4; ++j)
            if (j < s)
              es += emb_new2[(size_t)sNodes[m * 4 + j] * 256 + n];
          float val = acc[r] + qbn + fmaxf(es, 0.0f);
          atomicAdd(node_acc + (size_t)sTgt[m] * 256 + n, val);
        }
      }
    }
  }
}

// ---------------------------------------------------------------------------
// Kernel 3: out = relu(node_acc / deg) + residual
// ---------------------------------------------------------------------------
__global__ __launch_bounds__(256) void k_final(
    const float* __restrict__ node_acc, const float* __restrict__ residual,
    const int* __restrict__ node_degree, float* __restrict__ out, int N) {
  long long total = (long long)N * 256;
  for (long long i = (long long)blockIdx.x * blockDim.x + threadIdx.x;
       i < total; i += (long long)gridDim.x * blockDim.x) {
    int n = (int)(i >> 8);
    float deg = (float)node_degree[n];
    float v = node_acc[i] / deg;
    out[i] = fmaxf(v, 0.0f) + residual[i];
  }
}

// ---------------------------------------------------------------------------
// Host launcher
// ---------------------------------------------------------------------------
extern "C" void kernel_launch(void* const* d_in, const int* in_sizes, int n_in,
                              void* d_out, int out_size, void* d_ws,
                              size_t ws_size, hipStream_t stream) {
  const float* embedding  = (const float*)d_in[0];
  const float* global_emb = (const float*)d_in[1];
  const float* pW   = (const float*)d_in[2];
  const float* pb   = (const float*)d_in[3];
  const float* qW   = (const float*)d_in[4];
  const float* qb   = (const float*)d_in[5];
  const float* p2W1 = (const float*)d_in[6];
  const float* p2b1 = (const float*)d_in[7];
  const float* p2W2 = (const float*)d_in[8];
  const float* p2b2 = (const float*)d_in[9];
  const float* aW   = (const float*)d_in[10];
  const float* ab   = (const float*)d_in[11];
  const int* edge_nodes  = (const int*)d_in[12];
  // d_in[13] = edge_mask (boolean) -- recomputed as (slot < edge_size)
  const int* edge_size   = (const int*)d_in[14];
  const int* node_degree = (const int*)d_in[15];
  const int* inc_node    = (const int*)d_in[16];
  const int* inc_edge    = (const int*)d_in[17];
  const int* inc_slot    = (const int*)d_in[18];
  float* out = (float*)d_out;

  const int N   = in_sizes[0] / 256;
  const int NNZ = in_sizes[16];

  // --- workspace sub-allocation (256B aligned) ---
  char* ws = (char*)d_ws;
  size_t off = 0;
  auto alloc = [&](size_t bytes) -> void* {
    void* p = ws + off;
    off = (off + bytes + 255) & ~(size_t)255;
    return p;
  };
  unsigned short* embB = (unsigned short*)alloc((size_t)N * 256 * 2);
  unsigned short* aWt  = (unsigned short*)alloc((size_t)256 * 256 * 2);
  float*          aB   = (float*)alloc(256 * 4);
  unsigned short* pWt  = (unsigned short*)alloc((size_t)64 * 256 * 2);
  float*          pB   = (float*)alloc(64 * 4);
  unsigned short* h1Wt = (unsigned short*)alloc((size_t)1024 * 256 * 2);
  float*          hB   = (float*)alloc(1024 * 4);
  unsigned short* h2Wt = (unsigned short*)alloc((size_t)256 * 1024 * 2);
  float*          b2   = (float*)alloc(256 * 4);
  unsigned short* qWt  = (unsigned short*)alloc((size_t)256 * 64 * 2);
  float*          qB   = (float*)alloc(256 * 4);
  float* emb_new  = (float*)alloc((size_t)N * 64 * 4);
  float* emb_new2 = (float*)alloc((size_t)N * 256 * 4);
  float* residual = (float*)alloc((size_t)N * 256 * 4);
  float* node_acc = (float*)alloc((size_t)N * 256 * 4);
  (void)ws_size; (void)n_in; (void)out_size;

  // --- prep ---
  {
    long long n = (long long)N * 256;
    int g = (int)((n + 255) / 256);
    if (g > 4096) g = 4096;
    k_cvt_bf16<<<g, 256, 0, stream>>>(embedding, embB, n);
  }
  k_prep_w<<<256, 256, 0, stream>>>(aW, ab, aWt, aB, 256, 256, 1);
  k_prep_w<<<64, 256, 0, stream>>>(pW, pb, pWt, pB, 256, 64, 1);
  k_prep_w<<<1024, 256, 0, stream>>>(p2W1, p2b1, h1Wt, hB, 256, 1024, 1);
  k_prep_w<<<1024, 256, 0, stream>>>(p2W2, p2b2, h2Wt, b2, 1024, 256, 0);
  k_prep_w<<<64, 256, 0, stream>>>(qW, qb, qWt, qB, 64, 256, 0);
  {
    long long n = (long long)N * 256;
    int g = (int)((n + 255) / 256);
    if (g > 4096) g = 4096;
    k_zero<<<g, 256, 0, stream>>>(node_acc, n);
  }

  // --- main pipeline ---
  {
    size_t shmem = (size_t)(32 * 256 + 32 * 1024) * sizeof(unsigned short);
    k_node<<<(N + 31) / 32, 256, shmem, stream>>>(embB, aWt, aB, pWt, pB, h1Wt,
                                                  hB, h2Wt, b2, emb_new,
                                                  residual, emb_new2, N);
  }
  k_edge<<<(NNZ + 63) / 64, 256, 0, stream>>>(
      emb_new, emb_new2, qWt, qB, global_emb, edge_nodes, edge_size,
      node_degree, inc_node, inc_edge, inc_slot, node_acc, NNZ);
  {
    long long n = (long long)N * 256;
    int g = (int)((n + 255) / 256);
    if (g > 4096) g = 4096;
    k_final<<<g, 256, 0, stream>>>(node_acc, residual, node_degree, out, N);
  }
}
